// NaiveAttentionBlock_90151363543484
// MI455X (gfx1250) — matrix operand order repaired and checked by reading, not compiled, not run on hardware
//
#include <hip/hip_runtime.h>

// ---- types -----------------------------------------------------------------
typedef __attribute__((ext_vector_type(16))) __bf16         v16bf;
typedef __attribute__((ext_vector_type(16))) unsigned short v16us;
typedef __attribute__((ext_vector_type(8)))  unsigned short v8us;
typedef __attribute__((ext_vector_type(8)))  float          v8f;
typedef __attribute__((ext_vector_type(4)))  float          v4f;
typedef __attribute__((ext_vector_type(4)))  unsigned int   v4u;

#define HIDDEN   2048
#define HEADS    16
#define HDIM     128
#define SEQ      2048
#define BATCH    2
#define ROWS     (BATCH * SEQ)        // 4096
#define QKV_N    (3 * HIDDEN)         // 6144
#define SCALE_F  0.08838834764831845f // 1/sqrt(128)

// fp32 -> bf16 (round to nearest even), as raw u16
static __device__ __forceinline__ unsigned short f2bfu(float f) {
    unsigned u = __builtin_bit_cast(unsigned, f);
    u += 0x7FFFu + ((u >> 16) & 1u);
    return (unsigned short)(u >> 16);
}

static __device__ __forceinline__ v8f wmma_bf16(v16bf a, v16bf b, v8f c) {
    return __builtin_amdgcn_wmma_f32_16x16x32_bf16(
        /*neg_a=*/false, a, /*neg_b=*/false, b,
        /*c_mod=*/(short)0, c, /*reuse_a=*/false, /*reuse_b=*/false);
}

// A-fragment (16x32 bf16) from bf16 row-major memory (global or LDS).
// p (16B aligned) points at row + k0 + half*16; chunks at +0 and +8 elems.
// elements j -> K = half*16 + j
struct U32x8 { v4u a, b; };
static __device__ __forceinline__ v16bf load_a_bf16(const unsigned short* p) {
    U32x8 u;
    u.a = *(const v4u*)p;
    u.b = *(const v4u*)(p + 8);
    return __builtin_bit_cast(v16bf, u);
}

// B-fragment (32x16 bf16): 16 contiguous bf16 (32B aligned) at
// W[col] + k0 + half*16. elements j -> K = half*16 + j
static __device__ __forceinline__ v16bf load_b_bf16(const unsigned short* p) {
    return *(const v16bf*)p;
}

// ---------------------------------------------------------------------------
// Kernel 0: streaming fp32 -> bf16 conversion (one-time, bandwidth-bound).
// n must be a multiple of 8 * blockDim * gridDim coverage (exact grids used).
// ---------------------------------------------------------------------------
__global__ void __launch_bounds__(256)
cvt_f32_to_bf16(const float* __restrict__ in, unsigned short* __restrict__ out) {
    const size_t i = ((size_t)blockIdx.x * 256 + threadIdx.x) * 8;
    v4f a = *(const v4f*)(in + i);
    v4f b = *(const v4f*)(in + i + 4);
    v8us t;
#pragma unroll
    for (int j = 0; j < 4; ++j) { t[j] = f2bfu(a[j]); t[4 + j] = f2bfu(b[j]); }
    *(v4u*)(out + i) = __builtin_bit_cast(v4u, t);
}

// ---------------------------------------------------------------------------
// Kernel 1: QKV = Xb @ Wqkvb^T (both bf16); scatter into bf16 Q[b,h,s,d],
// K[b,h,s,d], Vt[b,h,d,s]. Wave tile 64x32 (4x2 wmma), 4 waves -> 128x64.
// ---------------------------------------------------------------------------
__global__ void __launch_bounds__(128)
qkv_gemm(const unsigned short* __restrict__ X, const unsigned short* __restrict__ W,
         unsigned short* __restrict__ qb, unsigned short* __restrict__ kb,
         unsigned short* __restrict__ vt) {
    const int lane = threadIdx.x & 31;
    const int half = lane >> 4;
    const int l16  = lane & 15;
    const int wave = threadIdx.x >> 5;
    const int wm = wave >> 1, wn = wave & 1;
    const int row0 = blockIdx.x * 128 + wm * 64;
    const int col0 = blockIdx.y * 64 + wn * 32;

    v8f acc[4][2] = {};
    for (int k0 = 0; k0 < HIDDEN; k0 += 32) {
        v16bf a[4], b[2];
#pragma unroll
        for (int mt = 0; mt < 4; ++mt)
            a[mt] = load_a_bf16(X + (size_t)(row0 + mt * 16 + l16) * HIDDEN + k0 + half * 16);
#pragma unroll
        for (int nt = 0; nt < 2; ++nt)
            b[nt] = load_b_bf16(W + (size_t)(col0 + nt * 16 + l16) * HIDDEN + k0 + half * 16);
#pragma unroll
        for (int mt = 0; mt < 4; ++mt)
#pragma unroll
            for (int nt = 0; nt < 2; ++nt)
                acc[mt][nt] = wmma_bf16(a[mt], b[nt], acc[mt][nt]);
    }

#pragma unroll
    for (int mt = 0; mt < 4; ++mt)
#pragma unroll
        for (int nt = 0; nt < 2; ++nt)
#pragma unroll
            for (int r = 0; r < 8; ++r) {
                const int row = row0 + mt * 16 + r + 8 * half;  // C/D: M = r + 8*half
                const int col = col0 + nt * 16 + l16;           //      N = lane&15
                const int bI = row >> 11, s = row & 2047;
                const int which = col >> 11, rr = col & 2047;
                const int h = rr >> 7, d = rr & 127;
                const unsigned short v = f2bfu(acc[mt][nt][r]);
                const size_t head = (size_t)(bI * HEADS + h);
                if (which == 0)      qb[(head * SEQ + s) * HDIM + d] = v;
                else if (which == 1) kb[(head * SEQ + s) * HDIM + d] = v;
                else                 vt[(head * HDIM + d) * SEQ + s] = v;
            }
}

// ---------------------------------------------------------------------------
// Kernel 2: flash attention. block = 4 waves; blockIdx.x = b*16+h,
// blockIdx.y = 64-query block; each wave owns 16 queries x d=128.
// ---------------------------------------------------------------------------
__global__ void __launch_bounds__(128)
attn_kernel(const unsigned short* __restrict__ qb,
            const unsigned short* __restrict__ kb,
            const unsigned short* __restrict__ vt,
            unsigned short* __restrict__ aout) {
    __shared__ __align__(32) unsigned short ldsP[4][16][64];  // per-wave P tile

    const int lane = threadIdx.x & 31;
    const int half = lane >> 4;
    const int l16  = lane & 15;
    const int wave = threadIdx.x >> 5;
    const int bh = blockIdx.x;
    const unsigned short* Qh = qb + (size_t)bh * SEQ * HDIM;
    const unsigned short* Kh = kb + (size_t)bh * SEQ * HDIM;
    const unsigned short* Vh = vt + (size_t)bh * HDIM * SEQ;
    const int qbase = blockIdx.y * 64 + wave * 16;

    // Q fragments: 4 K-slices of 32 over d=128 (resident for whole kernel)
    v16bf qf[4];
#pragma unroll
    for (int kd = 0; kd < 4; ++kd)
        qf[kd] = load_a_bf16(Qh + (size_t)(qbase + l16) * HDIM + kd * 32 + half * 16);

    v8f o[8] = {};
    float m[8], l[8];
#pragma unroll
    for (int r = 0; r < 8; ++r) { m[r] = -3.0e38f; l[r] = 0.0f; }

    for (int kbk = 0; kbk < SEQ; kbk += 64) {
        // S = Q K^T for 64 keys (4 N-tiles)
        v8f s[4] = {};
#pragma unroll
        for (int nt = 0; nt < 4; ++nt)
#pragma unroll
            for (int kd = 0; kd < 4; ++kd) {
                v16bf bfr = load_b_bf16(Kh + (size_t)(kbk + nt * 16 + l16) * HDIM +
                                        kd * 32 + half * 16);
                s[nt] = wmma_bf16(qf[kd], bfr, s[nt]);
            }

        // online softmax per row (row = 16 lanes within a wave half)
#pragma unroll
        for (int r = 0; r < 8; ++r) {
            float s0 = s[0][r] * SCALE_F, s1 = s[1][r] * SCALE_F;
            float s2 = s[2][r] * SCALE_F, s3 = s[3][r] * SCALE_F;
            float mx = fmaxf(fmaxf(s0, s1), fmaxf(s2, s3));
#pragma unroll
            for (int off = 8; off >= 1; off >>= 1) mx = fmaxf(mx, __shfl_xor(mx, off, 32));
            const float mnew  = fmaxf(m[r], mx);
            const float alpha = __expf(m[r] - mnew);
            const float p0 = __expf(s0 - mnew), p1 = __expf(s1 - mnew);
            const float p2 = __expf(s2 - mnew), p3 = __expf(s3 - mnew);
            float rs = p0 + p1 + p2 + p3;
#pragma unroll
            for (int off = 8; off >= 1; off >>= 1) rs += __shfl_xor(rs, off, 32);
            l[r] = l[r] * alpha + rs;
            m[r] = mnew;
#pragma unroll
            for (int dt = 0; dt < 8; ++dt) o[dt][r] *= alpha;
            const int M = r + 8 * half;
            ldsP[wave][M][ 0 + l16] = f2bfu(p0);
            ldsP[wave][M][16 + l16] = f2bfu(p1);
            ldsP[wave][M][32 + l16] = f2bfu(p2);
            ldsP[wave][M][48 + l16] = f2bfu(p3);
        }

        // O += P @ V  (V stored transposed: Vt[d][s])
#pragma unroll
        for (int dt = 0; dt < 8; ++dt)
#pragma unroll
            for (int ks = 0; ks < 2; ++ks) {
                v16bf pa = load_a_bf16(&ldsP[wave][l16][ks * 32 + half * 16]);
                v16bf vb = load_b_bf16(Vh + (size_t)(dt * 16 + l16) * SEQ +
                                       kbk + ks * 32 + half * 16);
                o[dt] = wmma_bf16(pa, vb, o[dt]);
            }
    }

    // normalize and store bf16 concat-head output [b, s, h*128 + d]
    const int bI = bh >> 4, h = bh & 15;
#pragma unroll
    for (int r = 0; r < 8; ++r) {
        const float inv = 1.0f / l[r];
        const int M = r + 8 * half;
        const size_t rowbase = ((size_t)(bI * SEQ + qbase + M)) * HIDDEN + h * HDIM;
#pragma unroll
        for (int dt = 0; dt < 8; ++dt)
            aout[rowbase + dt * 16 + l16] = f2bfu(o[dt][r] * inv);
    }
}

// ---------------------------------------------------------------------------
// Kernel 3: out = attn_out(bf16) @ Wprojb^T(bf16) -> fp32.
// Wave tile 64x32 (4x2 wmma), 4 waves -> 128x64 block tile.
// ---------------------------------------------------------------------------
__global__ void __launch_bounds__(128)
proj_gemm(const unsigned short* __restrict__ A, const unsigned short* __restrict__ W,
          float* __restrict__ out) {
    const int lane = threadIdx.x & 31;
    const int half = lane >> 4;
    const int l16  = lane & 15;
    const int wave = threadIdx.x >> 5;
    const int wm = wave >> 1, wn = wave & 1;
    const int row0 = blockIdx.x * 128 + wm * 64;
    const int col0 = blockIdx.y * 64 + wn * 32;

    v8f acc[4][2] = {};
    for (int k0 = 0; k0 < HIDDEN; k0 += 32) {
        v16bf a[4], b[2];
#pragma unroll
        for (int mt = 0; mt < 4; ++mt)
            a[mt] = load_a_bf16(A + (size_t)(row0 + mt * 16 + l16) * HIDDEN + k0 + half * 16);
#pragma unroll
        for (int nt = 0; nt < 2; ++nt)
            b[nt] = load_b_bf16(W + (size_t)(col0 + nt * 16 + l16) * HIDDEN + k0 + half * 16);
#pragma unroll
        for (int mt = 0; mt < 4; ++mt)
#pragma unroll
            for (int nt = 0; nt < 2; ++nt)
                acc[mt][nt] = wmma_bf16(a[mt], b[nt], acc[mt][nt]);
    }

#pragma unroll
    for (int mt = 0; mt < 4; ++mt)
#pragma unroll
        for (int nt = 0; nt < 2; ++nt)
#pragma unroll
            for (int r = 0; r < 8; ++r) {
                const int row = row0 + mt * 16 + r + 8 * half;
                const int col = col0 + nt * 16 + l16;
                out[(size_t)row * HIDDEN + col] = acc[mt][nt][r];
            }
}

// ---------------------------------------------------------------------------
extern "C" void kernel_launch(void* const* d_in, const int* in_sizes, int n_in,
                              void* d_out, int out_size, void* d_ws, size_t ws_size,
                              hipStream_t stream) {
    const float* x      = (const float*)d_in[0];   // [2,2048,2048]
    const float* w_qkv  = (const float*)d_in[1];   // [6144,2048]
    const float* w_proj = (const float*)d_in[2];   // [2048,2048]
    float* out = (float*)d_out;                    // [2,2048,2048] fp32

    const size_t N_X    = (size_t)ROWS * HIDDEN;    //  8,388,608
    const size_t N_WQKV = (size_t)QKV_N * HIDDEN;   // 12,582,912
    const size_t N_WPRJ = (size_t)HIDDEN * HIDDEN;  //  4,194,304
    const size_t HB     = (size_t)BATCH * HEADS * SEQ * HDIM;  // 8,388,608

    unsigned short* ws     = (unsigned short*)d_ws;
    unsigned short* xb     = ws;                         // x   bf16
    unsigned short* wqkvb  = xb + N_X;                   // w_qkv bf16
    unsigned short* wprojb = wqkvb + N_WQKV;             // w_proj bf16
    unsigned short* qb     = wprojb + N_WPRJ;            // Q  [b,h,s,d]
    unsigned short* kbuf   = qb + HB;                    // K  [b,h,s,d]
    unsigned short* vt     = kbuf + HB;                  // Vt [b,h,d,s]
    unsigned short* aout   = vt + HB;                    // attn out [b,s,2048]
    (void)in_sizes; (void)n_in; (void)out_size; (void)ws_size;

    cvt_f32_to_bf16<<<(unsigned)(N_X    / (8 * 256)), 256, 0, stream>>>(x,      xb);
    cvt_f32_to_bf16<<<(unsigned)(N_WQKV / (8 * 256)), 256, 0, stream>>>(w_qkv,  wqkvb);
    cvt_f32_to_bf16<<<(unsigned)(N_WPRJ / (8 * 256)), 256, 0, stream>>>(w_proj, wprojb);

    qkv_gemm<<<dim3(ROWS / 128, QKV_N / 64), 128, 0, stream>>>(xb, wqkvb, qb, kbuf, vt);
    attn_kernel<<<dim3(BATCH * HEADS, SEQ / 64), 128, 0, stream>>>(qb, kbuf, vt, aout);
    proj_gemm<<<dim3(ROWS / 128, HIDDEN / 64), 128, 0, stream>>>(aout, wprojb, out);
}
